// FDB_18751827214900
// MI455X (gfx1250) — matrix operand order repaired and checked by reading, not compiled, not run on hardware
//
#include <hip/hip_runtime.h>

#define B_ 16
#define C_ 64
#define H_ 256
#define W_ 256
#define NHW_ (B_*H_*W_)           // 1,048,576 reduction elements per channel
#define CUTOFF 90.0f
#define BN_EPS 1e-5f

#define TSTRIDE 72                // padded halo-tile stride in halves (144B = 36 banks)

typedef _Float16 half_t;
typedef __attribute__((ext_vector_type(16))) _Float16 v16h;
typedef __attribute__((ext_vector_type(8)))  _Float16 v8h;
typedef __attribute__((ext_vector_type(8)))  float    v8f;

// ---------------------------------------------------------------------------
// Kernel 1: freq = x * dm.  Writes f32 freq into out[:, C:2C] (NCHW) and an
// NHWC fp16 copy into workspace (channel-contiguous for the implicit GEMM).
// Grid: B*H*(W/64) blocks of 256 threads; each block does a 64ch x 64w tile.
// ---------------------------------------------------------------------------
__global__ void __launch_bounds__(256) freq_kernel(const float* __restrict__ x,
                                                   float* __restrict__ out,
                                                   half_t* __restrict__ freq_h) {
    __shared__ half_t lds[64 * 65];   // +1 pad to dodge bank conflicts
    int bid = blockIdx.x;             // ((b*H + h)*4 + wt)
    int wt  = bid & 3;
    int h   = (bid >> 2) & (H_ - 1);
    int b   = bid >> 10;
    int w0  = wt * 64;
    int t   = threadIdx.x;

#pragma unroll
    for (int i = 0; i < 16; ++i) {
        int lin = t + i * 256;
        int c   = lin >> 6;           // channel (slow)
        int wl  = lin & 63;           // w within tile (fast -> coalesced NCHW)
        int w   = w0 + wl;
        float xv = x[(((size_t)b * C_ + c) * H_ + h) * W_ + w];
        float u  = (float)(h - H_ / 2);
        float v  = (float)(w - W_ / 2);
        float d  = sqrtf(u * u + v * v);
        d = (d == 0.0f) ? 1e-6f : d;
        float r    = d * (1.0f / CUTOFF);
        float filt = 1.0f / (1.0f + r * r);      // Butterworth order 1
        float dm   = 0.8f * filt + 0.5f;
        float f    = xv * dm;
        out[(((size_t)b * (2 * C_) + (C_ + c)) * H_ + h) * W_ + w] = f;
        lds[c * 65 + wl] = (half_t)f;
    }
    __syncthreads();
#pragma unroll
    for (int i = 0; i < 16; ++i) {
        int lin = t + i * 256;
        int wl  = lin >> 6;           // w (slow)
        int c   = lin & 63;           // channel (fast -> coalesced NHWC)
        freq_h[(((size_t)b * H_ + h) * W_ + w0 + wl) * C_ + c] = lds[c * 65 + wl];
    }
}

// ---------------------------------------------------------------------------
// Kernel 2: repack conv_w [O][I][3][3] f32 -> [tap][O][I] fp16.
// ---------------------------------------------------------------------------
__global__ void wprep_kernel(const float* __restrict__ cw, half_t* __restrict__ wh) {
    int idx = blockIdx.x * 256 + threadIdx.x;
    if (idx >= 9 * 64 * 64) return;
    int tap = idx >> 12;              // /4096
    int o   = (idx >> 6) & 63;
    int i   = idx & 63;
    int ky  = tap / 3, kx = tap % 3;
    wh[idx] = (half_t)cw[((o * 64 + i) * 3 + ky) * 3 + kx];
}

// ---------------------------------------------------------------------------
// Kernel 3: implicit-GEMM 3x3 conv with v_wmma_f32_16x16x32_f16.
// One block = (b, h, 32-wide w tile); 8 waves each own 16ch x 16px C tile.
// Halo tile staged with CDNA5 global_load_async_to_lds (ASYNCcnt path),
// padded to a 144B column stride so B-fragment ds_load_b128 is conflict-free.
// ---------------------------------------------------------------------------
__global__ void __launch_bounds__(256) conv_kernel(const half_t* __restrict__ freq_h,
                                                   const half_t* __restrict__ wh,
                                                   const float* __restrict__ bias,
                                                   half_t* __restrict__ y_h) {
    __shared__ __align__(16) half_t tile[3 * 34 * TSTRIDE];
    int bid = blockIdx.x;             // ((b*H + h)*8 + wt)
    int wt  = bid & 7;
    int h   = (bid >> 3) & (H_ - 1);
    int b   = bid >> 11;
    int w0  = wt * 32;
    int t   = threadIdx.x;

    // Stage the 3-row x 34-col x 64-ch halo tile straight into LDS with the
    // async tensor/data path; zero-fill out-of-image halo chunks via ds store.
    unsigned lds_base = (unsigned)(uintptr_t)(void*)tile;
    for (int k = t; k < 816; k += 256) {
        int rc   = k >> 3;            // (row,col) pair 0..101
        int part = k & 7;             // which 8-half chunk of the 64 channels
        int row  = rc / 34;
        int col  = rc % 34;
        int gh   = h + row - 1;
        int gw   = w0 + col - 1;
        unsigned ldst = lds_base + (unsigned)(rc * TSTRIDE + part * 8) * 2;
        if ((unsigned)gh < (unsigned)H_ && (unsigned)gw < (unsigned)W_) {
            unsigned long long gsrc = (unsigned long long)(uintptr_t)
                (freq_h + (((size_t)b * H_ + gh) * W_ + gw) * C_ + part * 8);
            asm volatile("global_load_async_to_lds_b128 %0, %1, off"
                         :: "v"(ldst), "v"(gsrc) : "memory");
        } else {
            v8h z = {};
            *(v8h*)(tile + rc * TSTRIDE + part * 8) = z;
        }
    }
    asm volatile("s_wait_asynccnt 0x0" ::: "memory");
    __syncthreads();

    int wave = t >> 5;
    int lane = t & 31;
    int l    = lane & 15;
    int hi   = lane >> 4;             // 0 or 1 (half-wave)
    int m16  = (wave >> 1) * 16;      // output-channel tile base
    int nt   = (wave & 1) * 16;       // spatial tile base within the 32 pixels

    v8f acc = {};
#pragma unroll
    for (int tap = 0; tap < 9; ++tap) {
        int dy = tap / 3, dx = tap % 3;
        const half_t* bp = tile + ((dy * 34) + nt + l + dx) * TSTRIDE;
#pragma unroll
        for (int kc = 0; kc < 2; ++kc) {
            // A fragment (ISA 16-bit A 16x32 layout): lane<16 holds K 0-7,16-23;
            // lane>=16 holds K 8-15,24-31 of this 32-wide K chunk.
            const half_t* ap = wh + ((tap * 64) + m16 + l) * 64 + kc * 32 + hi * 8;
            v8h alo = *(const v8h*)(ap);
            v8h ahi = *(const v8h*)(ap + 16);
            v16h a;
#pragma unroll
            for (int j = 0; j < 8; ++j) { a[j] = alo[j]; a[8 + j] = ahi[j]; }
            // B fragment: lane holds column N=l, K 0-7,16-23 (lane<16) or
            // K 8-15,24-31 (lane>=16) -> two 16B conflict-free LDS loads.
            const half_t* bp2 = bp + kc * 32 + hi * 8;
            v8h blo = *(const v8h*)(bp2);
            v8h bhi = *(const v8h*)(bp2 + 16);
            v16h bfrag;
#pragma unroll
            for (int j = 0; j < 8; ++j) { bfrag[j] = blo[j]; bfrag[8 + j] = bhi[j]; }
            acc = __builtin_amdgcn_wmma_f32_16x16x32_f16(
                false, a, false, bfrag, (short)0, acc, false, false);
        }
    }

    // C layout: VGPR r -> channel m16 + 8*hi + r, pixel w0 + nt + l.
    // 8 consecutive channels == 16B contiguous NHWC store.
    int ch0 = m16 + hi * 8;
    v8h outv;
#pragma unroll
    for (int r = 0; r < 8; ++r) outv[r] = (half_t)(acc[r] + bias[ch0 + r]);
    *(v8h*)(y_h + (((size_t)b * H_ + h) * W_ + w0 + nt + l) * C_ + ch0) = outv;
}

// ---------------------------------------------------------------------------
// Kernel 4a/4b: per-channel sum & sumsq of y (NHWC) for BatchNorm.
// ---------------------------------------------------------------------------
__global__ void zero_stats(float* __restrict__ s) {
    if (threadIdx.x < 128) s[threadIdx.x] = 0.0f;
}

__global__ void __launch_bounds__(256) reduce_kernel(const half_t* __restrict__ y_h,
                                                     float* __restrict__ stats) {
    __shared__ float ssum[32][64];
    __shared__ float ssq[32][64];
    int t     = threadIdx.x;
    int rgrp  = t >> 3;               // 0..31: row group
    int chunk = t & 7;                // 8-channel chunk -> 16B vector loads
    size_t row0 = (size_t)blockIdx.x * 1024;   // 1024 blocks cover B*H*W rows
    float s[8], q[8];
#pragma unroll
    for (int j = 0; j < 8; ++j) { s[j] = 0.0f; q[j] = 0.0f; }
    for (int r = rgrp; r < 1024; r += 32) {
        v8h v = *(const v8h*)(y_h + (row0 + r) * 64 + chunk * 8);
#pragma unroll
        for (int j = 0; j < 8; ++j) {
            float f = (float)v[j];
            s[j] += f; q[j] += f * f;
        }
    }
#pragma unroll
    for (int j = 0; j < 8; ++j) {
        ssum[rgrp][chunk * 8 + j] = s[j];
        ssq[rgrp][chunk * 8 + j]  = q[j];
    }
    __syncthreads();
    if (t < 64) {
        float ts = 0.0f, tq = 0.0f;
#pragma unroll
        for (int g = 0; g < 32; ++g) { ts += ssum[g][t]; tq += ssq[g][t]; }
        atomicAdd(&stats[t], ts);
        atomicAdd(&stats[64 + t], tq);
    }
}

// ---------------------------------------------------------------------------
// Kernel 5: BN (training stats) + SiLU, NHWC fp16 -> NCHW f32 out[:, 0:C].
// ---------------------------------------------------------------------------
__global__ void __launch_bounds__(256) finalize_kernel(const half_t* __restrict__ y_h,
                                                       const float* __restrict__ stats,
                                                       const float* __restrict__ gamma,
                                                       const float* __restrict__ beta,
                                                       float* __restrict__ out) {
    __shared__ float lds[64 * 65];
    int bid = blockIdx.x;             // ((b*H + h)*4 + wt)
    int wt  = bid & 3;
    int h   = (bid >> 2) & (H_ - 1);
    int b   = bid >> 10;
    int w0  = wt * 64;
    int t   = threadIdx.x;
    const float inv_n = 1.0f / (float)NHW_;

#pragma unroll
    for (int i = 0; i < 16; ++i) {
        int lin = t + i * 256;
        int c   = lin & 63;           // channel fast -> coalesced NHWC read
        int wl  = lin >> 6;
        float sum  = stats[c];
        float sq   = stats[64 + c];
        float mean = sum * inv_n;
        float var  = sq * inv_n - mean * mean;
        float sc   = gamma[c] * rsqrtf(var + BN_EPS);
        float sh   = beta[c] - mean * sc;
        float v    = (float)y_h[(((size_t)b * H_ + h) * W_ + w0 + wl) * C_ + c];
        float yn   = v * sc + sh;
        float act  = yn / (1.0f + __expf(-yn));   // SiLU
        lds[c * 65 + wl] = act;
    }
    __syncthreads();
#pragma unroll
    for (int i = 0; i < 16; ++i) {
        int lin = t + i * 256;
        int wl  = lin & 63;           // w fast -> coalesced NCHW write
        int c   = lin >> 6;
        out[(((size_t)b * (2 * C_) + c) * H_ + h) * W_ + w0 + wl] = lds[c * 65 + wl];
    }
}

// ---------------------------------------------------------------------------
extern "C" void kernel_launch(void* const* d_in, const int* in_sizes, int n_in,
                              void* d_out, int out_size, void* d_ws, size_t ws_size,
                              hipStream_t stream) {
    (void)in_sizes; (void)n_in; (void)out_size; (void)ws_size;
    const float* x      = (const float*)d_in[0];
    const float* conv_w = (const float*)d_in[1];
    const float* conv_b = (const float*)d_in[2];
    const float* gamma  = (const float*)d_in[3];
    const float* beta   = (const float*)d_in[4];
    float* out = (float*)d_out;

    char* ws = (char*)d_ws;
    const size_t elems = (size_t)B_ * C_ * H_ * W_;          // 67,108,864
    half_t* freq_h = (half_t*)ws;                            // 134,217,728 B
    half_t* y_h    = (half_t*)(ws + elems * 2);              // 134,217,728 B
    half_t* w_h    = (half_t*)(ws + elems * 4);              // 73,728 B
    float*  stats  = (float*)(ws + elems * 4 + 73728);       // 512 B

    freq_kernel<<<B_ * H_ * (W_ / 64), 256, 0, stream>>>(x, out, freq_h);
    wprep_kernel<<<(9 * 64 * 64 + 255) / 256, 256, 0, stream>>>(conv_w, w_h);
    zero_stats<<<1, 128, 0, stream>>>(stats);
    conv_kernel<<<B_ * H_ * (W_ / 32), 256, 0, stream>>>(freq_h, w_h, conv_b, y_h);
    reduce_kernel<<<1024, 256, 0, stream>>>(y_h, stats);
    finalize_kernel<<<B_ * H_ * (W_ / 64), 256, 0, stream>>>(y_h, stats, gamma, beta, out);
}
